// CrossAttentionLayer_70978629534358
// MI455X (gfx1250) — matrix operand order repaired and checked by reading, not compiled, not run on hardware
//
#include <hip/hip_runtime.h>

#define DEVI __device__ __forceinline__

typedef __attribute__((ext_vector_type(16))) __bf16 v16bf;
typedef __attribute__((ext_vector_type(8)))  float  v8f;

static constexpr int Bn = 16, Sn = 2048, En = 1024, Hn = 4, DHn = 256, E3n = 3072;

union FragU { v16bf v; unsigned u[8]; };

DEVI unsigned short f2bf(float f) {
    union { float f; unsigned u; } c; c.f = f;
    unsigned r = c.u + 0x7fffu + ((c.u >> 16) & 1u);   // round-to-nearest-even
    return (unsigned short)(r >> 16);
}
DEVI unsigned pack2bf(float lo, float hi) {
    return (unsigned)f2bf(lo) | ((unsigned)f2bf(hi) << 16);
}

// A fragment: 16x32 bf16, lane row = lane&15, group g = lane>>4.
// halves 0..7  -> K = g*8 + 0..7 ; halves 8..15 -> K = 16 + g*8 + 0..7
DEVI v16bf load_a(const unsigned short* p /* row base + k0 */, int g) {
    FragU f;
    const unsigned* q0 = (const unsigned*)(p + g * 8);
    const unsigned* q1 = (const unsigned*)(p + 16 + g * 8);
    f.u[0] = q0[0]; f.u[1] = q0[1]; f.u[2] = q0[2]; f.u[3] = q0[3];
    f.u[4] = q1[0]; f.u[5] = q1[1]; f.u[6] = q1[2]; f.u[7] = q1[3];
    return f.v;
}
// B fragment: 32x16 bf16, lane col = lane&15, halves 0..7 -> K = g*8 + 0..7 ; halves 8..15 -> K = 16 + g*8 + 0..7
DEVI v16bf load_b(const unsigned short* p /* col-row base + k0 */, int g) {
    FragU f;
    const unsigned* q0 = (const unsigned*)(p + g * 8);
    const unsigned* q1 = (const unsigned*)(p + 16 + g * 8);
#pragma unroll
    for (int j = 0; j < 4; ++j) { f.u[j] = q0[j]; f.u[4 + j] = q1[j]; }
    return f.v;
}
DEVI v8f wmma_bf16(v16bf a, v16bf b, v8f c) {
    return __builtin_amdgcn_wmma_f32_16x16x32_bf16(false, a, false, b, (short)0, c, false, false);
}

// ---------------- f32 -> bf16 conversion ----------------
__global__ void k_cvt_bf16(const float* __restrict__ src, unsigned short* __restrict__ dst, int n) {
    int i = blockIdx.x * blockDim.x + threadIdx.x;
    int stride = gridDim.x * blockDim.x;
    for (; i < n; i += stride) dst[i] = f2bf(src[i]);
}

// ---------------- QKV projection GEMM (64x64 per wave) ----------------
// qkv[r,c] = sum_k xb[r,k] * wbi[c,k] + bias[c]; scatter into Q,K (row) and V^T.
__global__ __launch_bounds__(256) void k_gemm_qkv(
    const unsigned short* __restrict__ xb, const unsigned short* __restrict__ wbi,
    const float* __restrict__ bias,
    unsigned short* __restrict__ qb, unsigned short* __restrict__ kb,
    unsigned short* __restrict__ vtb) {
    const int tilesN = E3n / 64;                       // 48
    int wid  = blockIdx.x * (blockDim.x >> 5) + (threadIdx.x >> 5);
    int lane = threadIdx.x & 31;
    int g = lane >> 4, l16 = lane & 15;
    int tm = wid / tilesN, tn = wid % tilesN;
    int r0 = tm * 64, c0 = tn * 64;

    const unsigned short* arow[4];
    const unsigned short* brow[4];
#pragma unroll
    for (int i = 0; i < 4; ++i) {
        arow[i] = xb  + (size_t)(r0 + i * 16 + l16) * En;
        brow[i] = wbi + (size_t)(c0 + i * 16 + l16) * En;
    }
    v8f acc[4][4];
#pragma unroll
    for (int i = 0; i < 4; ++i)
#pragma unroll
        for (int j = 0; j < 4; ++j) acc[i][j] = (v8f){0.f,0.f,0.f,0.f,0.f,0.f,0.f,0.f};

    for (int k0 = 0; k0 < En; k0 += 32) {
        v16bf a[4];
#pragma unroll
        for (int i = 0; i < 4; ++i) a[i] = load_a(arow[i] + k0, g);
#pragma unroll
        for (int j = 0; j < 4; ++j) {
            v16bf bfr = load_b(brow[j] + k0, g);
#pragma unroll
            for (int i = 0; i < 4; ++i) acc[i][j] = wmma_bf16(a[i], bfr, acc[i][j]);
        }
    }

#pragma unroll
    for (int j = 0; j < 4; ++j) {
        int n = c0 + j * 16 + l16;
        float bn = bias[n];
        int which = n >> 10;              // 0=q 1=k 2=v (uniform per wave per j)
        int e = n & 1023, h = e >> 8, d = e & 255;
#pragma unroll
        for (int i = 0; i < 4; ++i) {
#pragma unroll
            for (int v = 0; v < 8; ++v) {
                int r = r0 + i * 16 + v + g * 8;
                int b = r >> 11, s = r & 2047;
                float val = acc[i][j][v] + bn;
                size_t bh = (size_t)(b * Hn + h);
                if (which == 0)      qb [(bh * Sn + s) * DHn + d] = f2bf(val * 0.0625f); // fold 1/sqrt(DH)
                else if (which == 1) kb [(bh * Sn + s) * DHn + d] = f2bf(val);
                else                 vtb[(bh * DHn + d) * Sn + s] = f2bf(val);
            }
        }
    }
}

// ---------------- Pass 1: softmax row stats (m, l) ----------------
__global__ __launch_bounds__(256) void k_attn_stats(
    const unsigned short* __restrict__ qb, const unsigned short* __restrict__ kb,
    const unsigned char* __restrict__ mask, float* __restrict__ ml) {
    __shared__ float smem[8][16][17];
    int w = threadIdx.x >> 5, lane = threadIdx.x & 31;
    int g = lane >> 4, l16 = lane & 15;
    int gw = blockIdx.x * 8 + w;                       // 8192 q-tiles total
    int bh = gw >> 7, qt = gw & 127;
    int b = bh >> 2;
    int s0 = qt * 16;

    const unsigned short* qrow = qb + ((size_t)bh * Sn + s0 + l16) * DHn;
    v16bf qa[8];
#pragma unroll
    for (int t = 0; t < 8; ++t) qa[t] = load_a(qrow + t * 32, g);

    const unsigned char* mrow = mask + (size_t)b * Sn;
    float mi = -3.0e38f, li = 0.f;

    for (int k0 = 0; k0 < Sn; k0 += 16) {
        const unsigned short* krow = kb + ((size_t)bh * Sn + k0 + l16) * DHn;
        v8f sc = {0.f, 0.f, 0.f, 0.f, 0.f, 0.f, 0.f, 0.f};
#pragma unroll
        for (int t = 0; t < 8; ++t) sc = wmma_bf16(qa[t], load_b(krow + t * 32, g), sc);
        __syncthreads();
#pragma unroll
        for (int v = 0; v < 8; ++v) smem[w][v + g * 8][l16] = sc[v];
        __syncthreads();
        if (lane < 16) {
            float sv[16], rmax = mi;
#pragma unroll
            for (int n = 0; n < 16; ++n) {
                float s = smem[w][lane][n];
                s = mrow[k0 + n] ? s : -1.0e9f;
                sv[n] = s;
                rmax = fmaxf(rmax, s);
            }
            float mold = mi; mi = rmax;
            li *= __expf(mold - mi);
#pragma unroll
            for (int n = 0; n < 16; ++n) li += __expf(sv[n] - mi);
        }
    }
    if (lane < 16) {
        size_t idx = (size_t)bh * Sn + s0 + lane;
        ml[idx * 2] = mi; ml[idx * 2 + 1] = li;
    }
}

// ---------------- Pass 2: P = softmax, colsum, ctx = P @ V ----------------
__global__ __launch_bounds__(256) void k_attn_ctx(
    const unsigned short* __restrict__ qb, const unsigned short* __restrict__ kb,
    const unsigned short* __restrict__ vtb, const unsigned char* __restrict__ mask,
    const float* __restrict__ ml, unsigned short* __restrict__ ctxb,
    float* __restrict__ colsum) {
    __shared__ float pmem[8][16][33];
    __shared__ float bcs[Sn];
    int w = threadIdx.x >> 5, lane = threadIdx.x & 31;
    int g = lane >> 4, l16 = lane & 15;
    int gw = blockIdx.x * 8 + w;
    int bh = gw >> 7, qt = gw & 127;                   // bh uniform within block
    int b = bh >> 2, h = bh & 3;
    int s0 = qt * 16;

    for (int i = threadIdx.x; i < Sn; i += 256) bcs[i] = 0.f;

    const unsigned short* qrow = qb + ((size_t)bh * Sn + s0 + l16) * DHn;
    v16bf qa[8];
#pragma unroll
    for (int t = 0; t < 8; ++t) qa[t] = load_a(qrow + t * 32, g);

    float mrv[8], ilv[8];
#pragma unroll
    for (int v = 0; v < 8; ++v) {
        size_t idx = (size_t)bh * Sn + s0 + v + g * 8;
        mrv[v] = ml[idx * 2];
        ilv[v] = 1.0f / ml[idx * 2 + 1];
    }

    v8f acc[16];
#pragma unroll
    for (int t = 0; t < 16; ++t) acc[t] = (v8f){0.f, 0.f, 0.f, 0.f, 0.f, 0.f, 0.f, 0.f};

    const unsigned char* mrow = mask + (size_t)b * Sn;
    const unsigned short* vbase = vtb + (size_t)bh * DHn * Sn;
    __syncthreads();

    for (int k0 = 0; k0 < Sn; k0 += 32) {
        const unsigned short* krow0 = kb + ((size_t)bh * Sn + k0 + l16) * DHn;
        const unsigned short* krow1 = krow0 + (size_t)16 * DHn;
        v8f sc0 = {0.f, 0.f, 0.f, 0.f, 0.f, 0.f, 0.f, 0.f};
        v8f sc1 = {0.f, 0.f, 0.f, 0.f, 0.f, 0.f, 0.f, 0.f};
#pragma unroll
        for (int t = 0; t < 8; ++t) sc0 = wmma_bf16(qa[t], load_b(krow0 + t * 32, g), sc0);
#pragma unroll
        for (int t = 0; t < 8; ++t) sc1 = wmma_bf16(qa[t], load_b(krow1 + t * 32, g), sc1);
        __syncthreads();
#pragma unroll
        for (int v = 0; v < 8; ++v) {
            int m = v + g * 8;
            float p0 = mrow[k0 + l16]      ? __expf(sc0[v] - mrv[v]) * ilv[v] : 0.f;
            float p1 = mrow[k0 + 16 + l16] ? __expf(sc1[v] - mrv[v]) * ilv[v] : 0.f;
            pmem[w][m][l16]      = p0;
            pmem[w][m][16 + l16] = p1;
        }
        __syncthreads();
        // column sums (per-block LDS accumulation)
        float cs = 0.f;
#pragma unroll
        for (int m = 0; m < 16; ++m) cs += pmem[w][m][lane];
        atomicAdd(&bcs[k0 + lane], cs);
        // pack P into A fragment (bf16)
        FragU pf;
#pragma unroll
        for (int j = 0; j < 4; ++j) {
            pf.u[j]     = pack2bf(pmem[w][l16][g * 8 + 2 * j],      pmem[w][l16][g * 8 + 2 * j + 1]);
            pf.u[4 + j] = pack2bf(pmem[w][l16][16 + g * 8 + 2 * j], pmem[w][l16][16 + g * 8 + 2 * j + 1]);
        }
        // ctx += P @ V : B lane col = d, K contiguous along keys (V^T rows)
#pragma unroll
        for (int t = 0; t < 16; ++t) {
            const unsigned short* brow = vbase + (size_t)(t * 16 + l16) * Sn + k0;
            acc[t] = wmma_bf16(pf.v, load_b(brow, g), acc[t]);
        }
    }

    // flush block colsum -> global (scaled by 1/H)
    __syncthreads();
    for (int i = threadIdx.x; i < Sn; i += 256)
        atomicAdd(&colsum[(size_t)b * Sn + i], bcs[i] * 0.25f);

    // store ctx (bf16, head-merged [B,S,E])
#pragma unroll
    for (int t = 0; t < 16; ++t) {
#pragma unroll
        for (int v = 0; v < 8; ++v) {
            int s = s0 + v + g * 8;
            int d = t * 16 + l16;
            ctxb[((size_t)(b * Sn + s)) * En + h * DHn + d] = f2bf(acc[t][v]);
        }
    }
}

// ---------------- Output projection GEMM (64x64 per wave) ----------------
__global__ __launch_bounds__(256) void k_gemm_out(
    const unsigned short* __restrict__ ctxb, const unsigned short* __restrict__ wbo,
    const float* __restrict__ bias, float* __restrict__ attn_out) {
    const int tilesN = En / 64;                        // 16
    int wid  = blockIdx.x * (blockDim.x >> 5) + (threadIdx.x >> 5);
    int lane = threadIdx.x & 31;
    int g = lane >> 4, l16 = lane & 15;
    int tm = wid / tilesN, tn = wid % tilesN;
    int r0 = tm * 64, c0 = tn * 64;

    const unsigned short* arow[4];
    const unsigned short* brow[4];
#pragma unroll
    for (int i = 0; i < 4; ++i) {
        arow[i] = ctxb + (size_t)(r0 + i * 16 + l16) * En;
        brow[i] = wbo  + (size_t)(c0 + i * 16 + l16) * En;
    }
    v8f acc[4][4];
#pragma unroll
    for (int i = 0; i < 4; ++i)
#pragma unroll
        for (int j = 0; j < 4; ++j) acc[i][j] = (v8f){0.f,0.f,0.f,0.f,0.f,0.f,0.f,0.f};

    for (int k0 = 0; k0 < En; k0 += 32) {
        v16bf a[4];
#pragma unroll
        for (int i = 0; i < 4; ++i) a[i] = load_a(arow[i] + k0, g);
#pragma unroll
        for (int j = 0; j < 4; ++j) {
            v16bf bfr = load_b(brow[j] + k0, g);
#pragma unroll
            for (int i = 0; i < 4; ++i) acc[i][j] = wmma_bf16(a[i], bfr, acc[i][j]);
        }
    }

#pragma unroll
    for (int j = 0; j < 4; ++j) {
        int n = c0 + j * 16 + l16;
        float bn = bias[n];
#pragma unroll
        for (int i = 0; i < 4; ++i) {
#pragma unroll
            for (int v = 0; v < 8; ++v) {
                int r = r0 + i * 16 + v + g * 8;
                attn_out[(size_t)r * En + n] = acc[i][j][v] + bn;
            }
        }
    }
}

// ---------------- Final reduction: out[b,e] = sum_k colsum[b,k]*attn_out[b,k,e] ----------------
__global__ __launch_bounds__(256) void k_final(
    const float* __restrict__ colsum, const float* __restrict__ attn_out,
    float* __restrict__ out) {
    int b = blockIdx.x;
    int e = blockIdx.y * 256 + threadIdx.x;
    float a = 0.f;
    const float* ao = attn_out + (size_t)b * Sn * En + e;
    const float* cs = colsum + (size_t)b * Sn;
    for (int k = 0; k < Sn; ++k) a += cs[k] * ao[(size_t)k * En];
    out[(size_t)b * En + e] = a;
}

extern "C" void kernel_launch(void* const* d_in, const int* in_sizes, int n_in,
                              void* d_out, int out_size, void* d_ws, size_t ws_size,
                              hipStream_t stream) {
    const float*         x    = (const float*)d_in[0];
    const unsigned char* mask = (const unsigned char*)d_in[1];
    const float*         wi   = (const float*)d_in[2];
    const float*         bi   = (const float*)d_in[3];
    const float*         wo   = (const float*)d_in[4];
    const float*         bo   = (const float*)d_in[5];
    float*               out  = (float*)d_out;

    char* ws = (char*)d_ws;
    size_t off = 0;
    auto alloc = [&](size_t bytes) -> void* {
        void* p = ws + off;
        off += (bytes + 255) & ~(size_t)255;
        return p;
    };
    unsigned short* xb   = (unsigned short*)alloc((size_t)Bn * Sn * En * 2);        // 64 MB
    unsigned short* wbi  = (unsigned short*)alloc((size_t)E3n * En * 2);            // 6 MB
    unsigned short* wbo  = (unsigned short*)alloc((size_t)En * En * 2);             // 2 MB
    unsigned short* qb   = (unsigned short*)alloc((size_t)Bn * Hn * Sn * DHn * 2);  // 64 MB
    unsigned short* kb   = (unsigned short*)alloc((size_t)Bn * Hn * Sn * DHn * 2);  // 64 MB
    unsigned short* vtb  = (unsigned short*)alloc((size_t)Bn * Hn * DHn * Sn * 2);  // 64 MB
    float*          ml   = (float*)alloc((size_t)Bn * Hn * Sn * 2 * 4);             // 1 MB
    unsigned short* ctxb = (unsigned short*)alloc((size_t)Bn * Sn * En * 2);        // 64 MB
    float*          aout = (float*)alloc((size_t)Bn * Sn * En * 4);                 // 128 MB
    float*          csum = (float*)alloc((size_t)Bn * Sn * 4);                      // 128 KB

    // 1) bf16 conversions
    k_cvt_bf16<<<2048, 256, 0, stream>>>(x,  xb,  Bn * Sn * En);
    k_cvt_bf16<<<2048, 256, 0, stream>>>(wi, wbi, E3n * En);
    k_cvt_bf16<<<1024, 256, 0, stream>>>(wo, wbo, En * En);

    // 2) QKV projection: 512x48 64x64-tiles, 8 waves/block
    k_gemm_qkv<<<(512 * 48) / 8, 256, 0, stream>>>(xb, wbi, bi, qb, kb, vtb);

    // 3) softmax row stats: 8192 q-tiles / 8 waves per block
    k_attn_stats<<<1024, 256, 0, stream>>>(qb, kb, mask, ml);

    // 4) zero column-sum accumulator, then P/ctx pass
    hipMemsetAsync(csum, 0, (size_t)Bn * Sn * 4, stream);
    k_attn_ctx<<<1024, 256, 0, stream>>>(qb, kb, vtb, mask, ml, ctxb, csum);

    // 5) output projection: 512x16 64x64-tiles, 8 waves/block
    k_gemm_out<<<(512 * 16) / 8, 256, 0, stream>>>(ctxb, wbo, bo, aout);

    // 6) final reduction to [B, E]
    k_final<<<dim3(Bn, En / 256), 256, 0, stream>>>(csum, aout, out);

    (void)in_sizes; (void)n_in; (void)out_size; (void)ws_size;
}
